// LabelGraphSubtypeHead_80796924772677
// MI455X (gfx1250) — matrix operand order repaired and checked by reading, not compiled, not run on hardware
//
#include <hip/hip_runtime.h>

#define B_TOTAL   16384
#define FDIM      2048
#define HID       256
#define NNODES    9
#define NH        (NNODES*HID)   // 2304
#define STEPS     4

typedef __attribute__((ext_vector_type(16))) __bf16 bf16x16;
typedef __attribute__((ext_vector_type(8)))  __bf16 bf16x8;
typedef __attribute__((ext_vector_type(8)))  float  f32x8;
typedef __attribute__((ext_vector_type(4)))  int    int4v;

#define GLOBAL_AS __attribute__((address_space(1)))
#define LDS_AS    __attribute__((address_space(3)))

__device__ __forceinline__ unsigned short f2bf(float f) {
  unsigned u = __builtin_bit_cast(unsigned, f);
  u += 0x7fffu + ((u >> 16) & 1u);          // round-to-nearest-even
  return (unsigned short)(u >> 16);
}
__device__ __forceinline__ float bf2f(unsigned short h) {
  return __builtin_bit_cast(float, ((unsigned)h) << 16);
}
__device__ __forceinline__ bf16x16 join8(bf16x8 lo, bf16x8 hi) {
  return __builtin_shufflevector(lo, hi, 0,1,2,3,4,5,6,7,8,9,10,11,12,13,14,15);
}
__device__ __forceinline__ f32x8 wmma_bf16(bf16x16 a, bf16x16 b, f32x8 c) {
  return __builtin_amdgcn_wmma_f32_16x16x32_bf16(false, a, false, b, (short)0, c, false, false);
}

// ---- async global->LDS 16-byte copy (ASYNCcnt path), with sync fallback ----
#if __has_builtin(__builtin_amdgcn_global_load_async_to_lds_b128)
#define HAS_ASYNC_LDS 1
#else
#define HAS_ASYNC_LDS 0
#endif

__device__ __forceinline__ void async_copy16(void* lds_dst, const void* g_src) {
#if HAS_ASYNC_LDS
  __builtin_amdgcn_global_load_async_to_lds_b128(
      (GLOBAL_AS int4v*)(int4v*)const_cast<void*>(g_src),
      (LDS_AS int4v*)(int4v*)lds_dst, 0, 0);
#else
  *(uint4*)lds_dst = *(const uint4*)g_src;
#endif
}

#if __has_builtin(__builtin_amdgcn_s_wait_asynccnt)
#define WAIT_ASYNC(n) __builtin_amdgcn_s_wait_asynccnt(n)
#elif HAS_ASYNC_LDS
#define WAIT_ASYNC(n) asm volatile("s_wait_asynccnt %0" ::"i"(n) : "memory")
#else
#define WAIT_ASYNC(n) ((void)0)
#endif

// A-fragment (16x32 bf16): lane<16 row=lane, K chunks {kk+0..7, kk+16..23};
// lane>=16 same row, K chunks {kk+8..15, kk+24..31}.
__device__ __forceinline__ bf16x16 load_frag_a(const unsigned short* base, int row0,
                                               int stride, int kk, int lane) {
  int half = lane >> 4, r = lane & 15;
  const unsigned short* p = base + (size_t)(row0 + r) * stride + kk + half * 8;
  bf16x8 lo = *(const bf16x8*)p;
  bf16x8 hi = *(const bf16x8*)(p + 16);
  return join8(lo, hi);
}
// B-fragment (32x16 bf16), W stored row-major [n_out][k]: lane<16 column n=lane
// K chunks {kk+0..7, kk+16..23}; lanes>=16 K chunks {kk+8..15, kk+24..31}.
__device__ __forceinline__ bf16x16 load_frag_b(const unsigned short* base, int row0,
                                               int stride, int kk, int lane) {
  int half = lane >> 4, r = lane & 15;
  const unsigned short* p = base + (size_t)(row0 + r) * stride + kk + half * 8;
  bf16x8 lo = *(const bf16x8*)p;
  bf16x8 hi = *(const bf16x8*)(p + 16);
  return join8(lo, hi);
}

// Row-normalized 9x9 adjacency, baked at compile time.
__device__ constexpr float A9[81] = {
  1.f/9,1.f/9,1.f/9,1.f/9,1.f/9,1.f/9,1.f/9,1.f/9,1.f/9,
  0.5f,0.5f,0,0,0,0,0,0,0,
  0.25f,0,0.25f,0,0,0,0,0.25f,0.25f,
  0.25f,0,0,0.25f,0,0,0,0.25f,0.25f,
  0.25f,0,0,0,0.25f,0,0,0.25f,0.25f,
  0.25f,0,0,0,0,0.25f,0,0.25f,0.25f,
  0.25f,0,0,0,0,0,0.25f,0.25f,0.25f,
  0.125f,0,0.125f,0.125f,0.125f,0.125f,0.125f,0.125f,0.125f,
  0.125f,0,0.125f,0.125f,0.125f,0.125f,0.125f,0.125f,0.125f,
};

// ---------------- Kernel 0a: fp32 -> bf16 bulk convert (float4 chunks) -----
__global__ __launch_bounds__(256) void cvt_bf16_kernel(const float* __restrict__ src,
                                                       unsigned short* __restrict__ dst,
                                                       int n4) {
  int i = blockIdx.x * blockDim.x + threadIdx.x;
  if (i < n4) {
    float4 v = *(const float4*)&src[(size_t)i * 4];
    unsigned long long pk =
        (unsigned long long)f2bf(v.x) | ((unsigned long long)f2bf(v.y) << 16) |
        ((unsigned long long)f2bf(v.z) << 32) | ((unsigned long long)f2bf(v.w) << 48);
    *(unsigned long long*)&dst[(size_t)i * 4] = pk;
  }
}

// ---------------- Kernel 0b: pack msg/upd weights fp32 -> bf16 -------------
__global__ void pack_weights_kernel(const float* __restrict__ msg_w,
                                    const float* __restrict__ upd_w,
                                    unsigned short* __restrict__ wm,
                                    unsigned short* __restrict__ w1,
                                    unsigned short* __restrict__ w2) {
  int i = blockIdx.x * blockDim.x + threadIdx.x;
  if (i < HID * HID) {
    wm[i] = f2bf(msg_w[i]);
    int n = i >> 8, k = i & 255;
    w1[i] = f2bf(upd_w[n * 2 * HID + k]);
    w2[i] = f2bf(upd_w[n * 2 * HID + HID + k]);
  }
}

// ---------------- Kernel 1: NS0 = conc @ f2n_w^T + f2n_b + node_enc --------
// 64(M) x 128(N) tile, 8 waves (2x4), each wave 32x32 = 2x2 WMMA tiles.
// Double-buffered: async global->LDS loads for panel k+1 overlap WMMA on k.
__global__ __launch_bounds__(256) void f2n_gemm_kernel(
    const unsigned short* __restrict__ concb,   // [16384][2048] bf16
    const unsigned short* __restrict__ f2nb,    // [2304][2048] bf16
    const float* __restrict__ f2n_b, const float* __restrict__ logits,
    const float* __restrict__ enc_w, const float* __restrict__ enc_b,
    unsigned short* __restrict__ ns) {
  __shared__ unsigned short As[2][64 * 72];     // [m][k], +8 pad
  __shared__ unsigned short Bs[2][128 * 72];    // [n][k], +8 pad
  const int tid  = threadIdx.x;
  const int lane = tid & 31;
  const int wave = tid >> 5;
  const int wm = wave >> 2, wn = wave & 3;
  const int m0 = blockIdx.x * 64;
  const int n0 = blockIdx.y * 128;

  // warm epilogue data while the pipeline fills (global_prefetch_b8)
  __builtin_prefetch(&logits[(size_t)(m0 + lane) * NNODES], 0, 1);

  // Exactly 6 async instructions per wave per K-panel (2 for A, 4 for B).
  auto issue_panel = [&](int buf, int k0) {
#pragma unroll
    for (int rr = 0; rr < 2; ++rr) {
      int c = tid + rr * 256;            // 0..511
      int row = c >> 3;                  // 0..63
      int k8 = (c & 7) << 3;             // 0..56
      async_copy16(&As[buf][row * 72 + k8],
                   &concb[(size_t)(m0 + row) * FDIM + k0 + k8]);
    }
#pragma unroll
    for (int rr = 0; rr < 4; ++rr) {
      int c = tid + rr * 256;            // 0..1023
      int row = c >> 3;                  // 0..127
      int k8 = (c & 7) << 3;
      async_copy16(&Bs[buf][row * 72 + k8],
                   &f2nb[(size_t)(n0 + row) * FDIM + k0 + k8]);
    }
  };

  f32x8 acc[2][2] = {};
  issue_panel(0, 0);
  int cur = 0;
  for (int k0 = 0; k0 < FDIM; k0 += 64) {
    if (k0 + 64 < FDIM) {
      issue_panel(cur ^ 1, k0 + 64);     // 12 outstanding ...
      WAIT_ASYNC(6);                     // ... retire the 6 for panel k0
    } else {
      WAIT_ASYNC(0);
    }
    __syncthreads();                     // panel k0 visible to all waves
#pragma unroll
    for (int ks = 0; ks < 2; ++ks) {
      int kk = ks * 32;
      bf16x16 a0 = load_frag_a(As[cur], wm * 32,      72, kk, lane);
      bf16x16 a1 = load_frag_a(As[cur], wm * 32 + 16, 72, kk, lane);
      bf16x16 b0 = load_frag_b(Bs[cur], wn * 32,      72, kk, lane);
      bf16x16 b1 = load_frag_b(Bs[cur], wn * 32 + 16, 72, kk, lane);
      acc[0][0] = wmma_bf16(a0, b0, acc[0][0]);
      acc[0][1] = wmma_bf16(a0, b1, acc[0][1]);
      acc[1][0] = wmma_bf16(a1, b0, acc[1][0]);
      acc[1][1] = wmma_bf16(a1, b1, acc[1][1]);
    }
    __syncthreads();                     // all reads done before buffer reuse
    cur ^= 1;
  }

  // epilogue: + f2n_b[col] + logits[b,node]*enc_w[h] + enc_b[h], store bf16
  const int half = lane >> 4, lr = lane & 15;
#pragma unroll
  for (int mi = 0; mi < 2; ++mi)
#pragma unroll
    for (int ni = 0; ni < 2; ++ni) {
      int col  = n0 + wn * 32 + ni * 16 + lr;
      int node = col >> 8;
      int h    = col & 255;
      float bias = f2n_b[col] + enc_b[h];
      float ew   = enc_w[h];
      int rowbase = m0 + wm * 32 + mi * 16 + half * 8;
#pragma unroll
      for (int r = 0; r < 8; ++r) {
        int row = rowbase + r;
        float v = acc[mi][ni][r] + bias + logits[row * NNODES + node] * ew;
        ns[(size_t)row * NH + col] = f2bf(v);
      }
    }
}

// ---------------- Kernel 2: fused 4-step message passing + heads -----------
// 1 block = 16 batch rows -> 144x256 state matrix in LDS; 9 waves, one 16-row
// m-tile each (in-place GEMM updates are wave-private -> no cross-wave hazard).
__global__ __launch_bounds__(288) void mp_kernel(
    const unsigned short* __restrict__ ns,
    const unsigned short* __restrict__ wmsg,
    const unsigned short* __restrict__ w1,
    const unsigned short* __restrict__ w2,
    const float* __restrict__ msg_b, const float* __restrict__ upd_b,
    const float* __restrict__ out_w, const float* __restrict__ out_b,
    float* __restrict__ out) {
  extern __shared__ unsigned short sm[];
  unsigned short* S = sm;                        // 144 x 264 node states
  unsigned short* T = sm + 144 * 264;            // 144 x 264 agg / msg buffer
  unsigned short* W = sm + 2 * 144 * 264;        // 256 x 264 weight stage
  const int tid  = threadIdx.x;
  const int lane = tid & 31;
  const int wave = tid >> 5;                     // 0..8
  const int half = lane >> 4, lr = lane & 15;
  const int b0 = blockIdx.x * 16;

  // async-load state tile (bf16, 16B chunks)
  for (int c = tid; c < 144 * 32; c += 288) {
    int row = c >> 5; int h8 = (c & 31) << 3;
    size_t g = (size_t)(b0 + row / 9) * NH + (row % 9) * HID + h8;
    async_copy16(&S[row * 264 + h8], &ns[g]);
  }
  WAIT_ASYNC(0);
  __syncthreads();

  auto stage_w = [&](const unsigned short* src) {
    for (int c = tid; c < 256 * 32; c += 288) {
      int row = c >> 5; int k8 = (c & 31) << 3;
      async_copy16(&W[row * 264 + k8], &src[row * HID + k8]);
    }
    WAIT_ASYNC(0);
  };

  for (int step = 0; step < STEPS; ++step) {
    // (1) aggregate states: T = A * S   (rows of A sum to 1 -> bias folds in)
    for (int e = tid; e < 16 * HID; e += 288) {
      int b = e >> 8, h = e & 255;
      float s[9];
#pragma unroll
      for (int j = 0; j < 9; ++j) s[j] = bf2f(S[(b * 9 + j) * 264 + h]);
#pragma unroll
      for (int i = 0; i < 9; ++i) {
        float a = 0.f;
#pragma unroll
        for (int j = 0; j < 9; ++j) a += A9[i * 9 + j] * s[j];
        T[(b * 9 + i) * 264 + h] = f2bf(a);
      }
    }
    __syncthreads();
    // (2) stage msg_w
    stage_w(wmsg);
    __syncthreads();
    // (3) msg GEMM in place: T(rows of this wave) = T @ msg_w^T + msg_b
    {
      f32x8 acc[16];
#pragma unroll
      for (int nt = 0; nt < 16; ++nt) {
        float bv = msg_b[nt * 16 + lr];
#pragma unroll
        for (int i = 0; i < 8; ++i) acc[nt][i] = bv;
      }
#pragma unroll
      for (int kt = 0; kt < 8; ++kt) {
        int kk = kt * 32;
        bf16x16 a = load_frag_a(T, wave * 16, 264, kk, lane);
#pragma unroll
        for (int nt = 0; nt < 16; ++nt) {
          bf16x16 b = load_frag_b(W, nt * 16, 264, kk, lane);
          acc[nt] = wmma_bf16(a, b, acc[nt]);
        }
      }
#pragma unroll
      for (int nt = 0; nt < 16; ++nt)
#pragma unroll
        for (int r = 0; r < 8; ++r)
          T[(wave * 16 + half * 8 + r) * 264 + nt * 16 + lr] = f2bf(acc[nt][r]);
    }
    __syncthreads();
    // (4) stage W1
    stage_w(w1);
    __syncthreads();
    // (5) upd pass 1: accU = S @ W1^T + upd_b
    f32x8 accU[16];
#pragma unroll
    for (int nt = 0; nt < 16; ++nt) {
      float bv = upd_b[nt * 16 + lr];
#pragma unroll
      for (int i = 0; i < 8; ++i) accU[nt][i] = bv;
    }
#pragma unroll
    for (int kt = 0; kt < 8; ++kt) {
      int kk = kt * 32;
      bf16x16 a = load_frag_a(S, wave * 16, 264, kk, lane);
#pragma unroll
      for (int nt = 0; nt < 16; ++nt) {
        bf16x16 b = load_frag_b(W, nt * 16, 264, kk, lane);
        accU[nt] = wmma_bf16(a, b, accU[nt]);
      }
    }
    __syncthreads();
    // (6) stage W2
    stage_w(w2);
    __syncthreads();
    // (7) upd pass 2: accU += T @ W2^T ; S = tanh(accU)
#pragma unroll
    for (int kt = 0; kt < 8; ++kt) {
      int kk = kt * 32;
      bf16x16 a = load_frag_a(T, wave * 16, 264, kk, lane);
#pragma unroll
      for (int nt = 0; nt < 16; ++nt) {
        bf16x16 b = load_frag_b(W, nt * 16, 264, kk, lane);
        accU[nt] = wmma_bf16(a, b, accU[nt]);
      }
    }
#pragma unroll
    for (int nt = 0; nt < 16; ++nt)
#pragma unroll
      for (int r = 0; r < 8; ++r)
        S[(wave * 16 + half * 8 + r) * 264 + nt * 16 + lr] = f2bf(tanhf(accU[nt][r]));
    __syncthreads();
  }

  // output heads: nodes 7 (chronic) and 8 (acute+chronic)
  if (tid < 32) {
    int b = tid >> 1, node = 7 + (tid & 1);
    float sum = out_b[0];
    for (int h = 0; h < HID; ++h)
      sum += bf2f(S[(b * 9 + node) * 264 + h]) * out_w[h];
    out[(size_t)(tid & 1) * B_TOTAL + b0 + b] = sum;
  }
}

extern "C" void kernel_launch(void* const* d_in, const int* in_sizes, int n_in,
                              void* d_out, int out_size, void* d_ws, size_t ws_size,
                              hipStream_t stream) {
  (void)in_sizes; (void)n_in; (void)out_size; (void)ws_size;
  const float* conc   = (const float*)d_in[0];
  const float* logits = (const float*)d_in[1];
  const float* enc_w  = (const float*)d_in[2];
  const float* enc_b  = (const float*)d_in[3];
  const float* f2n_w  = (const float*)d_in[4];
  const float* f2n_b  = (const float*)d_in[5];
  const float* msg_w  = (const float*)d_in[6];
  const float* msg_b  = (const float*)d_in[7];
  const float* upd_w  = (const float*)d_in[8];
  const float* upd_b  = (const float*)d_in[9];
  const float* out_w  = (const float*)d_in[10];
  const float* out_b  = (const float*)d_in[11];
  float* out = (float*)d_out;

  unsigned short* ns    = (unsigned short*)d_ws;              // 16384*2304 bf16
  unsigned short* concb = ns + (size_t)B_TOTAL * NH;          // 16384*2048 bf16
  unsigned short* f2nb  = concb + (size_t)B_TOTAL * FDIM;     // 2304*2048 bf16
  unsigned short* wm    = f2nb + (size_t)NH * FDIM;
  unsigned short* w1    = wm + HID * HID;
  unsigned short* w2    = w1 + HID * HID;

  {
    int n4 = B_TOTAL * FDIM / 4;
    cvt_bf16_kernel<<<n4 / 256, 256, 0, stream>>>(conc, concb, n4);
  }
  {
    int n4 = NH * FDIM / 4;
    cvt_bf16_kernel<<<n4 / 256, 256, 0, stream>>>(f2n_w, f2nb, n4);
  }
  pack_weights_kernel<<<(HID * HID + 255) / 256, 256, 0, stream>>>(msg_w, upd_w, wm, w1, w2);

  dim3 g1(B_TOTAL / 64, NH / 128);
  f2n_gemm_kernel<<<g1, 256, 0, stream>>>(concb, f2nb, f2n_b, logits, enc_w, enc_b, ns);

  size_t smem = (size_t)(2 * 144 * 264 + 256 * 264) * sizeof(unsigned short);  // 287232 B
  mp_kernel<<<B_TOTAL / 16, 288, smem, stream>>>(ns, wm, w1, w2, msg_b, upd_b,
                                                 out_w, out_b, out);
}
